// GraphSAGE_31628139167865
// MI455X (gfx1250) — compile-verified, matrix operand order checked by reading, not run on hardware
//
#include <hip/hip_runtime.h>

// ---------------------------------------------------------------------------
// GraphSAGE 3-layer forward for MI455X (gfx1250, wave32, WMMA).
// Aggregation is the roofline bottleneck (edge gather + segment-sum RMW);
// GEMMs run on v_wmma_f32_16x16x32_bf16 with fp32 accumulation.
// ---------------------------------------------------------------------------

typedef __attribute__((ext_vector_type(16))) __bf16 v16bf;
typedef __attribute__((ext_vector_type(8)))  float  v8f;

struct __attribute__((aligned(16))) U4 { unsigned int x, y, z, w; };
struct __attribute__((aligned(16))) F4 { float x, y, z, w; };

union V16 {
  v16bf v;
  U4 q[2];
  unsigned short s[16];
};

static __device__ __forceinline__ unsigned short f2bf(float f) {
  unsigned int u = __float_as_uint(f);
  u += 0x7FFFu + ((u >> 16) & 1u);            // round-to-nearest-even
  return (unsigned short)(u >> 16);
}

// ----------------------------- elementwise ---------------------------------

__global__ void zero_f32_kernel(float* __restrict__ p, long long n) {
  long long i = (long long)blockIdx.x * blockDim.x + threadIdx.x;
  if (i < n) p[i] = 0.0f;
}

__global__ void deg_count_kernel(const int* __restrict__ dst, int e,
                                 float* __restrict__ deg) {
  int i = blockIdx.x * blockDim.x + threadIdx.x;
  if (i < e) atomicAdd(&deg[dst[i]], 1.0f);
}

__global__ void deg_inv_kernel(float* __restrict__ deg, int n) {
  int i = blockIdx.x * blockDim.x + threadIdx.x;
  if (i < n) { float d = deg[i]; deg[i] = 1.0f / fmaxf(d, 1.0f); }
}

__global__ void cvt_bf16_kernel(const float* __restrict__ in,
                                unsigned short* __restrict__ out, long long n) {
  long long i = (long long)blockIdx.x * blockDim.x + threadIdx.x;
  if (i < n) out[i] = f2bf(in[i]);
}

// Pack weight matrix W[din x dout] (row-major fp32) into per-lane WMMA-B
// order: P[((kt*NT + nt)*32 + lane)*16 + j] where element j of lane L is
// K = kt*32 + 16*(j/8) + 8*(L/16) + (j%8), col = nt*16 + L%16.
__global__ void pack_w_kernel(const float* __restrict__ W,
                              unsigned short* __restrict__ P,
                              int din, int dout) {
  int i = blockIdx.x * blockDim.x + threadIdx.x;
  int total = din * dout;
  if (i >= total) return;
  int j    = i & 15;
  int lane = (i >> 4) & 31;
  int tile = i >> 9;
  int NT   = dout >> 4;
  int nt   = tile % NT;
  int kt   = tile / NT;
  int kh   = lane >> 4;
  int K    = kt * 32 + ((j < 8) ? (kh * 8 + j) : (16 + kh * 8 + (j - 8)));
  int col  = nt * 16 + (lane & 15);
  P[i] = f2bf(W[(long long)K * dout + col]);
}

// ------------------------------ aggregation --------------------------------
// One thread = one (edge, 8-feature chunk). With din=256 a wave32 covers one
// edge's full row: coalesced 512B gather (bf16), 8 fp32 atomic adds into the
// L2-resident accumulator.
__global__ void scatter_add_kernel(const unsigned short* __restrict__ h,
                                   const int* __restrict__ src,
                                   const int* __restrict__ dst,
                                   long long total, int cshift, int din,
                                   float* __restrict__ agg) {
  long long tid = (long long)blockIdx.x * blockDim.x + threadIdx.x;
  if (tid >= total) return;
  int c       = (int)(tid & ((1 << cshift) - 1));
  long long e = tid >> cshift;
  int s = src[e], d = dst[e];
  U4 q = *(const U4*)(h + (long long)s * din + c * 8);
  float* out = agg + (long long)d * din + c * 8;
  atomicAdd(out + 0, __uint_as_float(q.x << 16));
  atomicAdd(out + 1, __uint_as_float(q.x & 0xFFFF0000u));
  atomicAdd(out + 2, __uint_as_float(q.y << 16));
  atomicAdd(out + 3, __uint_as_float(q.y & 0xFFFF0000u));
  atomicAdd(out + 4, __uint_as_float(q.z << 16));
  atomicAdd(out + 5, __uint_as_float(q.z & 0xFFFF0000u));
  atomicAdd(out + 6, __uint_as_float(q.w << 16));
  atomicAdd(out + 7, __uint_as_float(q.w & 0xFFFF0000u));
}

// ------------------------------ fused GEMM ---------------------------------
// out = relu(h @ Ws + (agg*inv) @ Wn + b). One wave computes a 16x64 output
// strip (4 WMMA accumulator tiles), K-loop over both operand pairs.
template <bool OUT_BF16>
__global__ __launch_bounds__(32)
void sage_gemm_kernel(const unsigned short* __restrict__ hbf,
                      const float* __restrict__ agg,
                      const float* __restrict__ inv,
                      const unsigned short* __restrict__ WsP,
                      const unsigned short* __restrict__ WnP,
                      const float* __restrict__ bias,
                      void* __restrict__ outp,
                      int din, int dout) {
  const int lane = (int)(threadIdx.x & 31u);
  const int m    = lane & 15;
  const int kh   = lane >> 4;
  const int rt   = blockIdx.x;
  const int cg   = blockIdx.y;
  const int NT   = dout >> 4;
  const int nk   = din >> 5;
  const long long row = (long long)rt * 16 + m;

  v8f acc[4] = {};

  // --- self path: h @ Ws (bf16 direct loads) ---
  {
    const unsigned short* arow = hbf + row * din;
    for (int kt = 0; kt < nk; ++kt) {
      V16 A;
      A.q[0] = *(const U4*)(arow + kt * 32 + kh * 8);
      A.q[1] = *(const U4*)(arow + kt * 32 + 16 + kh * 8);
#pragma unroll
      for (int t = 0; t < 4; ++t) {
        const unsigned short* bp =
            WsP + (((long long)(kt * NT + cg * 4 + t) << 5) + lane) * 16;
        V16 B;
        B.q[0] = *(const U4*)bp;
        B.q[1] = *(const U4*)(bp + 8);
        acc[t] = __builtin_amdgcn_wmma_f32_16x16x32_bf16(
            false, A.v, false, B.v, (short)0, acc[t], false, false);
      }
    }
  }

  // --- neighbor path: (agg * inv_deg) @ Wn (fp32 load, fused scale+cvt) ---
  {
    const float s = inv[row];
    const float* arow = agg + row * din;
    for (int kt = 0; kt < nk; ++kt) {
      F4 f0 = *(const F4*)(arow + kt * 32 + kh * 8);
      F4 f1 = *(const F4*)(arow + kt * 32 + kh * 8 + 4);
      F4 f2 = *(const F4*)(arow + kt * 32 + 16 + kh * 8);
      F4 f3 = *(const F4*)(arow + kt * 32 + 16 + kh * 8 + 4);
      V16 A;
      A.s[0]  = f2bf(s * f0.x);  A.s[1]  = f2bf(s * f0.y);
      A.s[2]  = f2bf(s * f0.z);  A.s[3]  = f2bf(s * f0.w);
      A.s[4]  = f2bf(s * f1.x);  A.s[5]  = f2bf(s * f1.y);
      A.s[6]  = f2bf(s * f1.z);  A.s[7]  = f2bf(s * f1.w);
      A.s[8]  = f2bf(s * f2.x);  A.s[9]  = f2bf(s * f2.y);
      A.s[10] = f2bf(s * f2.z);  A.s[11] = f2bf(s * f2.w);
      A.s[12] = f2bf(s * f3.x);  A.s[13] = f2bf(s * f3.y);
      A.s[14] = f2bf(s * f3.z);  A.s[15] = f2bf(s * f3.w);
#pragma unroll
      for (int t = 0; t < 4; ++t) {
        const unsigned short* bp =
            WnP + (((long long)(kt * NT + cg * 4 + t) << 5) + lane) * 16;
        V16 B;
        B.q[0] = *(const U4*)bp;
        B.q[1] = *(const U4*)(bp + 8);
        acc[t] = __builtin_amdgcn_wmma_f32_16x16x32_bf16(
            false, A.v, false, B.v, (short)0, acc[t], false, false);
      }
    }
  }

  // --- epilogue: bias + relu, store (bf16 intermediate / fp32 final) ---
#pragma unroll
  for (int t = 0; t < 4; ++t) {
    const int col = cg * 64 + t * 16 + m;
    const float bb = bias[col];
#pragma unroll
    for (int r = 0; r < 8; ++r) {
      float v = acc[t][r] + bb;
      v = v > 0.0f ? v : 0.0f;
      const long long orow = (long long)rt * 16 + r + kh * 8;
      if (OUT_BF16)
        ((unsigned short*)outp)[orow * dout + col] = f2bf(v);
      else
        ((float*)outp)[orow * dout + col] = v;
    }
  }
}

// ------------------------------- launcher ----------------------------------

extern "C" void kernel_launch(void* const* d_in, const int* in_sizes, int n_in,
                              void* d_out, int out_size, void* d_ws,
                              size_t ws_size, hipStream_t stream) {
  (void)in_sizes; (void)n_in; (void)out_size; (void)ws_size;

  const int N = 100000, E = 1600000;
  const int D0 = 128, DH = 256;

  const float* feat = (const float*)d_in[0];
  const int*   src  = (const int*)d_in[1];
  const int*   dst  = (const int*)d_in[2];
  const float* Ws0  = (const float*)d_in[3];
  const float* Wn0  = (const float*)d_in[4];
  const float* b0   = (const float*)d_in[5];
  const float* Ws1  = (const float*)d_in[6];
  const float* Wn1  = (const float*)d_in[7];
  const float* b1   = (const float*)d_in[8];
  const float* Ws2  = (const float*)d_in[9];
  const float* Wn2  = (const float*)d_in[10];
  const float* b2   = (const float*)d_in[11];

  char* ws = (char*)d_ws;
  size_t off = 0;
  auto take = [&](size_t bytes) {
    char* p = ws + off;
    off = (off + bytes + 255) & ~(size_t)255;
    return p;
  };
  float*          inv = (float*)take((size_t)N * 4);
  unsigned short* hA  = (unsigned short*)take((size_t)N * DH * 2);
  unsigned short* hB  = (unsigned short*)take((size_t)N * DH * 2);
  float*          agg = (float*)take((size_t)N * DH * 4);
  unsigned short* P0s = (unsigned short*)take((size_t)D0 * DH * 2);
  unsigned short* P0n = (unsigned short*)take((size_t)D0 * DH * 2);
  unsigned short* P1s = (unsigned short*)take((size_t)DH * DH * 2);
  unsigned short* P1n = (unsigned short*)take((size_t)DH * DH * 2);
  unsigned short* P2s = (unsigned short*)take((size_t)DH * D0 * 2);
  unsigned short* P2n = (unsigned short*)take((size_t)DH * D0 * 2);

  auto cdiv = [](long long a, long long b) { return (unsigned)((a + b - 1) / b); };

  // degrees -> 1/max(deg,1)
  zero_f32_kernel<<<cdiv(N, 256), 256, 0, stream>>>(inv, N);
  deg_count_kernel<<<cdiv(E, 256), 256, 0, stream>>>(dst, E, inv);
  deg_inv_kernel<<<cdiv(N, 256), 256, 0, stream>>>(inv, N);

  // features -> bf16 plane (hB)
  cvt_bf16_kernel<<<cdiv((long long)N * D0, 256), 256, 0, stream>>>(
      feat, hB, (long long)N * D0);

  // pack weights into WMMA-B order
  pack_w_kernel<<<cdiv(D0 * DH, 256), 256, 0, stream>>>(Ws0, P0s, D0, DH);
  pack_w_kernel<<<cdiv(D0 * DH, 256), 256, 0, stream>>>(Wn0, P0n, D0, DH);
  pack_w_kernel<<<cdiv(DH * DH, 256), 256, 0, stream>>>(Ws1, P1s, DH, DH);
  pack_w_kernel<<<cdiv(DH * DH, 256), 256, 0, stream>>>(Wn1, P1n, DH, DH);
  pack_w_kernel<<<cdiv(DH * D0, 256), 256, 0, stream>>>(Ws2, P2s, DH, D0);
  pack_w_kernel<<<cdiv(DH * D0, 256), 256, 0, stream>>>(Wn2, P2n, DH, D0);

  // ---- layer 0: din=128 -> dout=256, out bf16 (hA) ----
  zero_f32_kernel<<<cdiv((long long)N * D0, 256), 256, 0, stream>>>(
      agg, (long long)N * D0);
  scatter_add_kernel<<<cdiv((long long)E * (D0 / 8), 256), 256, 0, stream>>>(
      hB, src, dst, (long long)E * (D0 / 8), 4, D0, agg);
  sage_gemm_kernel<true><<<dim3(N / 16, DH / 64), 32, 0, stream>>>(
      hB, agg, inv, P0s, P0n, b0, hA, D0, DH);

  // ---- layer 1: din=256 -> dout=256, out bf16 (hB) ----
  zero_f32_kernel<<<cdiv((long long)N * DH, 256), 256, 0, stream>>>(
      agg, (long long)N * DH);
  scatter_add_kernel<<<cdiv((long long)E * (DH / 8), 256), 256, 0, stream>>>(
      hA, src, dst, (long long)E * (DH / 8), 5, DH, agg);
  sage_gemm_kernel<true><<<dim3(N / 16, DH / 64), 32, 0, stream>>>(
      hA, agg, inv, P1s, P1n, b1, hB, DH, DH);

  // ---- layer 2: din=256 -> dout=128, out fp32 (d_out) ----
  zero_f32_kernel<<<cdiv((long long)N * DH, 256), 256, 0, stream>>>(
      agg, (long long)N * DH);
  scatter_add_kernel<<<cdiv((long long)E * (DH / 8), 256), 256, 0, stream>>>(
      hB, src, dst, (long long)E * (DH / 8), 5, DH, agg);
  sage_gemm_kernel<false><<<dim3(N / 16, D0 / 64), 32, 0, stream>>>(
      hB, agg, inv, P2s, P2n, b2, d_out, DH, D0);
}
